// GraphSAGE_13039520710795
// MI455X (gfx1250) — matrix operand order repaired and checked by reading, not compiled, not run on hardware
//
#include <hip/hip_runtime.h>

#define N_NODES   100000
#define N_EDGES   1600000
#define FEATS     128

typedef __attribute__((ext_vector_type(16))) __bf16 v16bf;
typedef __attribute__((ext_vector_type(8)))  __bf16 bf16x8;
typedef __attribute__((ext_vector_type(8)))  float  v8f;

// f32 -> bf16 bits (round-to-nearest-even), returned in low 16 bits
__device__ __forceinline__ unsigned bfbits(float f) {
    union { float f; unsigned u; } in; in.f = f;
    unsigned r = in.u + 0x7FFFu + ((in.u >> 16) & 1u);
    return r >> 16;
}
__device__ __forceinline__ unsigned pack2(float lo, float hi) {
    return (bfbits(hi) << 16) | bfbits(lo);
}

// ---------------------------------------------------------------- zero fill
__global__ void zero_kernel(float* __restrict__ p, int n4) {
    int i = blockIdx.x * blockDim.x + threadIdx.x;
    if (i < n4) reinterpret_cast<float4*>(p)[i] = make_float4(0.f, 0.f, 0.f, 0.f);
}

// ------------------------------------------------- gather + scatter-add
// One wave per edge: lane handles 4 consecutive floats of the 128-float row.
// Coalesced 512B row gather; native global_atomic_add_f32 scatter.
__global__ void scatter_kernel(const float* __restrict__ h,
                               const int*   __restrict__ src,
                               const int*   __restrict__ dst,
                               float*       __restrict__ agg) {
    int t    = blockIdx.x * blockDim.x + threadIdx.x;
    int e    = t >> 5;
    int lane = t & 31;
    if (e >= N_EDGES) return;
    int s = src[e];
    int d = dst[e];
    float4 v = reinterpret_cast<const float4*>(h + (long long)s * FEATS)[lane];
    float* out = agg + (long long)d * FEATS + lane * 4;
    __hip_atomic_fetch_add(out + 0, v.x, __ATOMIC_RELAXED, __HIP_MEMORY_SCOPE_AGENT);
    __hip_atomic_fetch_add(out + 1, v.y, __ATOMIC_RELAXED, __HIP_MEMORY_SCOPE_AGENT);
    __hip_atomic_fetch_add(out + 2, v.z, __ATOMIC_RELAXED, __HIP_MEMORY_SCOPE_AGENT);
    __hip_atomic_fetch_add(out + 3, v.w, __ATOMIC_RELAXED, __HIP_MEMORY_SCOPE_AGENT);
}

// ------------------------------------------------- f32 -> bf16 bulk convert
// One pass over the L2-resident agg matrix; 8 elements per thread.
__global__ void tobf16_kernel(const float* __restrict__ in,
                              unsigned*    __restrict__ out, int n8) {
    int i = blockIdx.x * blockDim.x + threadIdx.x;
    if (i >= n8) return;
    float4 a = reinterpret_cast<const float4*>(in)[2 * i];
    float4 b = reinterpret_cast<const float4*>(in)[2 * i + 1];
    uint4 o;
    o.x = pack2(a.x, a.y);
    o.y = pack2(a.z, a.w);
    o.z = pack2(b.x, b.y);
    o.w = pack2(b.z, b.w);
    reinterpret_cast<uint4*>(out)[i] = o;
}

// ------------------------------------------------- pack W into WMMA B layout
// Wp layout: [colTile][chunk][lane][16 bf16 contiguous]. Lane = N column
// (lanes 16-31 offset K by 16); element 2v/2v+1 = packed K pair {2v, 2v+1}.
template <int DOUT>
__global__ void packW_kernel(const float* __restrict__ W,
                             unsigned*    __restrict__ Wp) {
    constexpr int COL_TILES = DOUT / 16;
    int t = blockIdx.x * blockDim.x + threadIdx.x;
    if (t >= COL_TILES * 4 * 32) return;
    int lane    = t & 31;
    int ch      = (t >> 5) & 3;
    int colTile = t >> 7;
    int col     = colTile * 16 + (lane & 15);
    int kbase   = ch * 32 + (lane >> 4) * 16;
    unsigned* out = Wp + (size_t)t * 8;
#pragma unroll
    for (int v = 0; v < 8; ++v) {
        out[v] = pack2(W[(kbase + 2 * v)     * DOUT + col],
                       W[(kbase + 2 * v + 1) * DOUT + col]);
    }
}

// ------------------------------------------------- WMMA GEMM + bias + ReLU
// out[N, DOUT] = relu(Abf[N,128] @ W[128,DOUT] + bias), bf16 inputs, f32 acc.
// One wave = one 16x16 tile, 4x v_wmma_f32_16x16x32_bf16. Inner loop is
// pure b128 loads + WMMA (all conversions hoisted out).
template <int DOUT>
__global__ void gemm_kernel(const unsigned short* __restrict__ Abf, // [N,128] bf16
                            const unsigned*       __restrict__ Wp,  // packed bf16
                            const float*          __restrict__ bias,
                            float*                __restrict__ out) {
    constexpr int COL_TILES = DOUT / 16;
    constexpr int ROW_SUB   = 8 / COL_TILES;

    const int wave = threadIdx.x >> 5;
    const int lane = threadIdx.x & 31;
    const int rowTile = blockIdx.x * ROW_SUB + wave / COL_TILES;
    const int colTile = wave % COL_TILES;

    const int m  = lane & 15;
    const int hi = lane >> 4;
    const int col = colTile * 16 + m;

    // A: lane's row, segments of 8 bf16 (16B each)
    const bf16x8* arow =
        reinterpret_cast<const bf16x8*>(Abf + (size_t)(rowTile * 16 + m) * 128);
    // B: packed so each lane's 16 bf16 are contiguous (32B)
    const bf16x8* wp = reinterpret_cast<const bf16x8*>(Wp);

    v8f c = {};
#pragma unroll
    for (int ch = 0; ch < 4; ++ch) {
        // A tile: K {koff..koff+15}, koff = hi*16
        bf16x8 a0 = arow[ch * 4 + hi * 2];
        bf16x8 a1 = arow[ch * 4 + hi * 2 + 1];
        v16bf a = __builtin_shufflevector(a0, a1,
                      0, 1, 2, 3, 4, 5, 6, 7, 8, 9, 10, 11, 12, 13, 14, 15);

        int bidx = ((colTile * 4 + ch) * 32 + lane) * 2;
        bf16x8 b0 = wp[bidx];
        bf16x8 b1 = wp[bidx + 1];
        v16bf b = __builtin_shufflevector(b0, b1,
                      0, 1, 2, 3, 4, 5, 6, 7, 8, 9, 10, 11, 12, 13, 14, 15);

        c = __builtin_amdgcn_wmma_f32_16x16x32_bf16(
                false, a, false, b, (short)0, c, false, false);
    }

    const float bs    = bias[col];
    const int   rbase = rowTile * 16 + hi * 8;
#pragma unroll
    for (int r = 0; r < 8; ++r) {
        float v = c[r] + bs;
        out[(long long)(rbase + r) * DOUT + col] = v > 0.f ? v : 0.f;
    }
}

// ---------------------------------------------------------------- launch
extern "C" void kernel_launch(void* const* d_in, const int* in_sizes, int n_in,
                              void* d_out, int out_size, void* d_ws, size_t ws_size,
                              hipStream_t stream) {
    const float* feat = (const float*)d_in[0];
    const int*   src  = (const int*)  d_in[1];
    const int*   dst  = (const int*)  d_in[2];
    const float* W1   = (const float*)d_in[3];
    const float* b1   = (const float*)d_in[4];
    const float* W2   = (const float*)d_in[5];
    const float* b2   = (const float*)d_in[6];
    const float* W3   = (const float*)d_in[7];
    const float* b3   = (const float*)d_in[8];

    const size_t NF = (size_t)N_NODES * FEATS;
    float*          agg   = (float*)d_ws;
    float*          h1    = agg + NF;
    float*          h2    = h1 + NF;
    unsigned short* aggbf = (unsigned short*)(h2 + NF);   // NF bf16
    unsigned*       Wp    = (unsigned*)(aggbf + NF);      // up to 128*128 bf16

    const int n4 = (int)(NF / 4);
    const int n8 = (int)(NF / 8);
    dim3 blk(256);
    dim3 zgrid((n4 + 255) / 256);
    dim3 cgrid((n8 + 255) / 256);
    dim3 sgrid((N_EDGES * 32) / 256);

    // ---- layer 1
    zero_kernel<<<zgrid, blk, 0, stream>>>(agg, n4);
    scatter_kernel<<<sgrid, blk, 0, stream>>>(feat, src, dst, agg);
    tobf16_kernel<<<cgrid, blk, 0, stream>>>(agg, (unsigned*)aggbf, n8);
    packW_kernel<128><<<4, blk, 0, stream>>>(W1, Wp);
    gemm_kernel<128><<<N_NODES / 16, blk, 0, stream>>>(aggbf, Wp, b1, h1);

    // ---- layer 2
    zero_kernel<<<zgrid, blk, 0, stream>>>(agg, n4);
    scatter_kernel<<<sgrid, blk, 0, stream>>>(h1, src, dst, agg);
    tobf16_kernel<<<cgrid, blk, 0, stream>>>(agg, (unsigned*)aggbf, n8);
    packW_kernel<128><<<4, blk, 0, stream>>>(W2, Wp);
    gemm_kernel<128><<<N_NODES / 16, blk, 0, stream>>>(aggbf, Wp, b2, h2);

    // ---- layer 3 (DOUT = 64)
    zero_kernel<<<zgrid, blk, 0, stream>>>(agg, n4);
    scatter_kernel<<<sgrid, blk, 0, stream>>>(h2, src, dst, agg);
    tobf16_kernel<<<cgrid, blk, 0, stream>>>(agg, (unsigned*)aggbf, n8);
    packW_kernel<64><<<2, blk, 0, stream>>>(W3, Wp);
    gemm_kernel<64><<<N_NODES / 32, blk, 0, stream>>>(aggbf, Wp, b3, (float*)d_out);
}
